// STU_962072674972
// MI455X (gfx1250) — matrix operand order repaired and checked by reading, not compiled, not run on hardware
//
#include <hip/hip_runtime.h>
#include <hip/hip_bf16.h>
#include <stddef.h>

#define BATCH 4
#define SEQ   2048
#define DIM_  1024
#define KF    24
#define SHALF 1024   // SEQ/2: even-lag half-rate conv length

typedef __attribute__((ext_vector_type(16))) __bf16 v16bf;
typedef __attribute__((ext_vector_type(8)))  __bf16 v8bf;
typedef __attribute__((ext_vector_type(8)))  float  v8f;
typedef int vint4 __attribute__((vector_size(16)));   // matches builtin param

#if defined(__has_builtin)
#if __has_builtin(__builtin_amdgcn_global_load_async_to_lds_b128) && \
    __has_builtin(__builtin_amdgcn_s_wait_asynccnt)
#define STU_ASYNC 1
#endif
#endif

static __device__ __forceinline__ unsigned short bf_hi_bits(float f) {
  union { float f; unsigned u; } c; c.f = f; return (unsigned short)(c.u >> 16);
}
static __device__ __forceinline__ float bf_back(unsigned short s) {
  union { unsigned u; float f; } c; c.u = ((unsigned)s) << 16; return c.f;
}

// 16-byte global->LDS copy: async on CDNA5 (ASYNCcnt), sync fallback otherwise.
static __device__ __forceinline__ void cp16(const unsigned short* gsrc,
                                            unsigned short* lds) {
#if defined(STU_ASYNC)
  __builtin_amdgcn_global_load_async_to_lds_b128(
      (__attribute__((address_space(1))) vint4*)(unsigned long long)(const void*)gsrc,
      (__attribute__((address_space(3))) vint4*)lds, 0, 0);
#else
  *(uint4*)lds = *(const uint4*)gsrc;
#endif
}

// ---------------------------------------------------------------------------
// Kernel 0a: split X (fp32) into bf16 hi/lo pair, row-major (8192 x 1024).
// ---------------------------------------------------------------------------
__global__ __launch_bounds__(256) void stu_split_x(
    const float* __restrict__ X, unsigned short* __restrict__ Xhi,
    unsigned short* __restrict__ Xlo) {
  const size_t base = ((size_t)blockIdx.x * 256 + threadIdx.x) * 4;
  const float4 v = *(const float4*)(X + base);
  const float f[4] = {v.x, v.y, v.z, v.w};
  unsigned long long ph = 0, pl = 0;
#pragma unroll
  for (int i = 0; i < 4; ++i) {
    const unsigned short h = bf_hi_bits(f[i]);
    const unsigned short l = bf_hi_bits(f[i] - bf_back(h));
    ph |= (unsigned long long)h << (16 * i);
    pl |= (unsigned long long)l << (16 * i);
  }
  *(unsigned long long*)(Xhi + base) = ph;
  *(unsigned long long*)(Xlo + base) = pl;
}

// ---------------------------------------------------------------------------
// Kernel 0b: split + transpose M_inputs -> Mt[n][k] bf16 hi/lo (1024 x 1024),
// so B-fragments become k-contiguous 32B runs.
// ---------------------------------------------------------------------------
__global__ __launch_bounds__(256) void stu_split_mt(
    const float* __restrict__ M, unsigned short* __restrict__ Mthi,
    unsigned short* __restrict__ Mtlo) {
  __shared__ float tile[32][33];
  const int tid = threadIdx.x;
  const int nx  = tid & 31;
  const int ky0 = tid >> 5;                 // 0..7
  const int n0 = blockIdx.x * 32;
  const int k0 = blockIdx.y * 32;
#pragma unroll
  for (int r = 0; r < 4; ++r) {
    const int kk = ky0 + 8 * r;
    tile[kk][nx] = M[(size_t)(k0 + kk) * DIM_ + n0 + nx];
  }
  __syncthreads();
#pragma unroll
  for (int r = 0; r < 4; ++r) {
    const int nn = ky0 + 8 * r;
    const float v = tile[nx][nn];           // = M[k0+nx][n0+nn]
    const unsigned short h = bf_hi_bits(v);
    const unsigned short l = bf_hi_bits(v - bf_back(h));
    Mthi[(size_t)(n0 + nn) * DIM_ + k0 + nx] = h;
    Mtlo[(size_t)(n0 + nn) * DIM_ + k0 + nx] = l;
  }
}

// ---------------------------------------------------------------------------
// Kernel 1: h2[j][d] = 2 * (filters[2j,:] @ M_filters[:,d]),  j in [0,1024)
// ---------------------------------------------------------------------------
__global__ __launch_bounds__(256) void stu_filter_proj(
    const float* __restrict__ filters, const float* __restrict__ Mf,
    float* __restrict__ h2) {
  const int idx = blockIdx.x * 256 + threadIdx.x;
  const int j = idx >> 10;
  const int d = idx & (DIM_ - 1);
  const float* frow = filters + (size_t)(2 * j) * KF;
  float acc = 0.f;
#pragma unroll
  for (int k = 0; k < KF; ++k)
    acc = fmaf(frow[k], Mf[k * DIM_ + d], acc);
  h2[(size_t)j * DIM_ + d] = 2.0f * acc;
}

// ---------------------------------------------------------------------------
// Kernel 2: x_proj = x @ M_inputs via bf16x3 WMMA, pre-split operands,
// double-buffered async-to-LDS pipeline (ASYNCcnt), 128x64 tile / 8 waves.
// ---------------------------------------------------------------------------
#define TM 128
#define TN 64
#define TK 32
#define LDAB 40   // bf16 row stride: 80B (16B-aligned chunks, bank-spread)
#define NT  (DIM_ / TK)

__global__ __launch_bounds__(256) void stu_input_proj(
    const unsigned short* __restrict__ Xhi, const unsigned short* __restrict__ Xlo,
    const unsigned short* __restrict__ Mthi, const unsigned short* __restrict__ Mtlo,
    float* __restrict__ XP) {
  __shared__ __align__(16) unsigned short Ah[2][TM][LDAB];
  __shared__ __align__(16) unsigned short Al[2][TM][LDAB];
  __shared__ __align__(16) unsigned short Bh[2][TN][LDAB];
  __shared__ __align__(16) unsigned short Bl[2][TN][LDAB];

  const int tid  = threadIdx.x;
  const int lane = tid & 31;
  const int wave = tid >> 5;
  const int wm = wave >> 1;            // 0..3
  const int wn = wave & 1;             // 0..1
  const int rowBase = wm * 32;
  const int colBase = wn * 32;
  const int r0 = blockIdx.x * TM;
  const int c0 = blockIdx.y * TN;
  const int l15  = lane & 15;
  const int hi16 = (lane >> 4) & 1;
  const int kbA = hi16 ? 16 : 0;       // A 16x32 bf16 layout (k-contiguous, matches B)
  const int kbB = hi16 ? 16 : 0;       // B 32x16 bf16 layout

  // Per-thread staging coordinates: 6 x 16B copies per tile.
  const int am  = (tid + 0)   >> 2, akc = ((tid + 0)   & 3) * 8;
  const int am2 = (tid + 256) >> 2, akc2 = ((tid + 256) & 3) * 8;
  const int bn  = tid >> 2,         bkc = (tid & 3) * 8;

  v8f zf = {};
  v8f acc[2][2];
#pragma unroll
  for (int a = 0; a < 2; ++a)
#pragma unroll
    for (int b = 0; b < 2; ++b) acc[a][b] = zf;

  // Issue tile 0.
  {
    const int kk = 0;
    cp16(Xhi + (size_t)(r0 + am) * DIM_ + kk + akc,  &Ah[0][am][akc]);
    cp16(Xlo + (size_t)(r0 + am) * DIM_ + kk + akc,  &Al[0][am][akc]);
    cp16(Xhi + (size_t)(r0 + am2) * DIM_ + kk + akc2, &Ah[0][am2][akc2]);
    cp16(Xlo + (size_t)(r0 + am2) * DIM_ + kk + akc2, &Al[0][am2][akc2]);
    cp16(Mthi + (size_t)(c0 + bn) * DIM_ + kk + bkc, &Bh[0][bn][bkc]);
    cp16(Mtlo + (size_t)(c0 + bn) * DIM_ + kk + bkc, &Bl[0][bn][bkc]);
  }

  for (int t = 0; t < NT; ++t) {
    const int cur = t & 1;
    if (t + 1 < NT) {                  // prefetch next tile into other buffer
      const int nb = cur ^ 1;
      const int kk = (t + 1) * TK;
      cp16(Xhi + (size_t)(r0 + am) * DIM_ + kk + akc,  &Ah[nb][am][akc]);
      cp16(Xlo + (size_t)(r0 + am) * DIM_ + kk + akc,  &Al[nb][am][akc]);
      cp16(Xhi + (size_t)(r0 + am2) * DIM_ + kk + akc2, &Ah[nb][am2][akc2]);
      cp16(Xlo + (size_t)(r0 + am2) * DIM_ + kk + akc2, &Al[nb][am2][akc2]);
      cp16(Mthi + (size_t)(c0 + bn) * DIM_ + kk + bkc, &Bh[nb][bn][bkc]);
      cp16(Mtlo + (size_t)(c0 + bn) * DIM_ + kk + bkc, &Bl[nb][bn][bkc]);
    }
#if defined(STU_ASYNC)
    if (t + 1 < NT) __builtin_amdgcn_s_wait_asynccnt(6);  // current tile landed
    else            __builtin_amdgcn_s_wait_asynccnt(0);
#endif
    __syncthreads();                   // publish current tile to all waves

    // Fragment loads: two contiguous 16B runs, k = 16*half + s (same order for A and B).
    v16bf ahi[2], alo[2], bhi[2], blo[2];
#pragma unroll
    for (int rt = 0; rt < 2; ++rt) {
      const int row = rowBase + rt * 16 + l15;
      const v8bf h0 = *(const v8bf*)&Ah[cur][row][kbA];
      const v8bf h1 = *(const v8bf*)&Ah[cur][row][kbA + 8];
      const v8bf l0 = *(const v8bf*)&Al[cur][row][kbA];
      const v8bf l1 = *(const v8bf*)&Al[cur][row][kbA + 8];
      ahi[rt] = __builtin_shufflevector(h0, h1, 0,1,2,3,4,5,6,7,8,9,10,11,12,13,14,15);
      alo[rt] = __builtin_shufflevector(l0, l1, 0,1,2,3,4,5,6,7,8,9,10,11,12,13,14,15);
    }
#pragma unroll
    for (int ct = 0; ct < 2; ++ct) {
      const int col = colBase + ct * 16 + l15;
      const v8bf h0 = *(const v8bf*)&Bh[cur][col][kbB];
      const v8bf h1 = *(const v8bf*)&Bh[cur][col][kbB + 8];
      const v8bf l0 = *(const v8bf*)&Bl[cur][col][kbB];
      const v8bf l1 = *(const v8bf*)&Bl[cur][col][kbB + 8];
      bhi[ct] = __builtin_shufflevector(h0, h1, 0,1,2,3,4,5,6,7,8,9,10,11,12,13,14,15);
      blo[ct] = __builtin_shufflevector(l0, l1, 0,1,2,3,4,5,6,7,8,9,10,11,12,13,14,15);
    }

    // bf16x3: hi*hi + hi*lo + lo*hi
#pragma unroll
    for (int rt = 0; rt < 2; ++rt)
#pragma unroll
      for (int ct = 0; ct < 2; ++ct) {
        acc[rt][ct] = __builtin_amdgcn_wmma_f32_16x16x32_bf16(
            false, ahi[rt], false, bhi[ct], (short)0, acc[rt][ct], false, false);
        acc[rt][ct] = __builtin_amdgcn_wmma_f32_16x16x32_bf16(
            false, ahi[rt], false, blo[ct], (short)0, acc[rt][ct], false, false);
        acc[rt][ct] = __builtin_amdgcn_wmma_f32_16x16x32_bf16(
            false, alo[rt], false, bhi[ct], (short)0, acc[rt][ct], false, false);
      }
    __syncthreads();                   // all waves done reading 'cur'
  }

  // C/D layout: VGPR i -> M=i (lanes 0-15) / M=8+i (lanes 16-31), N = lane&15
#pragma unroll
  for (int rt = 0; rt < 2; ++rt)
#pragma unroll
    for (int ct = 0; ct < 2; ++ct) {
      const int colg = c0 + colBase + ct * 16 + l15;
#pragma unroll
      for (int i = 0; i < 8; ++i) {
        const int rowg = r0 + rowBase + rt * 16 + i + hi16 * 8;
        XP[(size_t)rowg * DIM_ + colg] = acc[rt][ct][i];
      }
    }
}

// ---------------------------------------------------------------------------
// Kernel 3: depthwise causal half-rate conv (fp32 VALU; per-channel Toeplitz
// cannot share a WMMA B operand across channels). Working set is L2-resident.
//   out[b, 2u+p, d] = sum_{j<=u} h2[j,d] * x_proj[b, 2(u-j)+p, d]
// ---------------------------------------------------------------------------
#define DC 64
#define UC 64
#define VC 64

__global__ __launch_bounds__(256) void stu_conv(
    const float* __restrict__ XP, const float* __restrict__ H2,
    float* __restrict__ OUT) {
  __shared__ float Xs[VC][DC + 1];
  __shared__ float Hs[UC + VC][DC + 1];
  const int tid = threadIdx.x;
  const int dl  = tid & 63;
  const int ug  = tid >> 6;
  const int u0  = blockIdx.x * UC;
  const int d0  = blockIdx.y * DC;
  const int b   = blockIdx.z >> 1;
  const int p   = blockIdx.z & 1;

  float acc[16];
#pragma unroll
  for (int i = 0; i < 16; ++i) acc[i] = 0.f;

  for (int v0 = 0; v0 <= u0; v0 += VC) {
#pragma unroll
    for (int i = 0; i < 16; ++i) {
      const int q  = tid + 256 * i;
      const int vl = q >> 6, dd = q & 63;
      Xs[vl][dd] = XP[((size_t)b * SEQ + 2 * (v0 + vl) + p) * DIM_ + d0 + dd];
    }
    const int jlo = u0 - v0 - (VC - 1);
    for (int i = 0; i < 32; ++i) {
      const int q = tid + 256 * i;
      if (q < (UC + VC - 1) * DC) {
        const int jl = q >> 6, dd = q & 63;
        const int j  = jlo + jl;
        Hs[jl][dd] = (j >= 0) ? H2[(size_t)j * DIM_ + d0 + dd] : 0.f;
      }
    }
    // Prefetch next chunk (lowers to global_prefetch_b8 on gfx1250).
    if (v0 + VC <= u0) {
      __builtin_prefetch(XP + ((size_t)b * SEQ + 2 * (v0 + VC) + p) * DIM_ + d0 + dl, 0, 0);
      if (jlo - VC >= 0)
        __builtin_prefetch(H2 + (size_t)(jlo - VC) * DIM_ + d0 + dl, 0, 0);
    }
    __syncthreads();

#pragma unroll 4
    for (int vv = 0; vv < VC; ++vv) {
      const float xv = Xs[vv][dl];
      const int jb = ug * 16 - vv + (VC - 1);
#pragma unroll
      for (int i = 0; i < 16; ++i)
        acc[i] = fmaf(Hs[jb + i][dl], xv, acc[i]);
    }
    __syncthreads();
  }

#pragma unroll
  for (int i = 0; i < 16; ++i) {
    const int u = u0 + ug * 16 + i;
    OUT[((size_t)b * SEQ + 2 * u + p) * DIM_ + d0 + dl] = acc[i];
  }
}

// ---------------------------------------------------------------------------
extern "C" void kernel_launch(void* const* d_in, const int* in_sizes, int n_in,
                              void* d_out, int out_size, void* d_ws, size_t ws_size,
                              hipStream_t stream) {
  (void)in_sizes; (void)n_in; (void)out_size; (void)ws_size;
  const float* x       = (const float*)d_in[0];   // (4, 2048, 1024)
  const float* filters = (const float*)d_in[1];   // (2048, 24)
  const float* Mi      = (const float*)d_in[2];   // (1024, 1024)
  const float* Mf      = (const float*)d_in[3];   // (24, 1024)
  float* out = (float*)d_out;

  // Workspace layout (75.5 MB total)
  float* xp = (float*)d_ws;                                   // 8192*1024 f32
  float* h2 = xp + (size_t)BATCH * SEQ * DIM_;                // 1024*1024 f32
  unsigned short* Xhi  = (unsigned short*)(h2 + (size_t)SHALF * DIM_);
  unsigned short* Xlo  = Xhi  + (size_t)BATCH * SEQ * DIM_;
  unsigned short* Mthi = Xlo  + (size_t)BATCH * SEQ * DIM_;
  unsigned short* Mtlo = Mthi + (size_t)DIM_ * DIM_;

  stu_split_x<<<(BATCH * SEQ * DIM_) / (256 * 4), 256, 0, stream>>>(x, Xhi, Xlo);
  stu_split_mt<<<dim3(DIM_ / 32, DIM_ / 32), 256, 0, stream>>>(Mi, Mthi, Mtlo);
  stu_filter_proj<<<(SHALF * DIM_) / 256, 256, 0, stream>>>(filters, Mf, h2);
  stu_input_proj<<<dim3((BATCH * SEQ) / TM, DIM_ / TN), 256, 0, stream>>>(
      Xhi, Xlo, Mthi, Mtlo, xp);
  stu_conv<<<dim3(SHALF / UC, DIM_ / DC, BATCH * 2), 256, 0, stream>>>(xp, h2, out);
}
